// UnslothGptOssExperts_32753420599349
// MI455X (gfx1250) — matrix operand order repaired and checked by reading, not compiled, not run on hardware
//
#include <hip/hip_runtime.h>
#include <hip/hip_bf16.h>

// ---------------- problem constants (match reference) ----------------
constexpr int E = 8;
constexpr int H = 2048;
constexpr int D = 2048;
constexpr int T = 2048;          // B*S = 4*512
constexpr float ALPHA = 1.702f;
constexpr float LIMIT = 7.0f;

constexpr int KB_A  = H / 32;    // 64 K-blocks for gate_up GEMM
constexpr int KB_B  = D / 32;    // 64 K-blocks for down GEMM
constexpr int NDSUB = D / 16;    // 128 d-subtiles
constexpr int NHSUB = H / 16;    // 128 h-subtiles

// workspace layout (dword offsets)
constexpr size_t W1P_OFF   = 0;                                    // packed gate_up_w frags
constexpr size_t W1P_DW    = (size_t)E * KB_A * NDSUB * 2 * 256;   // 33,554,432 dw (134 MB)
constexpr size_t W2P_OFF   = W1P_OFF + W1P_DW;                     // packed down_w frags
constexpr size_t W2P_DW    = (size_t)E * KB_B * NHSUB * 256;       // 16,777,216 dw (67 MB)
constexpr size_t XB_OFF    = W2P_OFF + W2P_DW;                     // x in bf16 row-major
constexpr size_t XB_DW     = (size_t)T * H / 2;                    // 2,097,152 dw (8 MB)
constexpr size_t FUSED_OFF = XB_OFF + XB_DW;                       // fused [E,T,D] bf16
// total ~264 MB of d_ws

// ---------------- WMMA types / helpers ----------------
typedef __bf16    v16bf __attribute__((ext_vector_type(16)));
typedef float     v8f   __attribute__((ext_vector_type(8)));
typedef float     v2f   __attribute__((ext_vector_type(2)));
typedef unsigned  v8u   __attribute__((ext_vector_type(8)));

__device__ __forceinline__ unsigned pack_bf16x2(float a, float b) {
    unsigned ua = __float_as_uint(a);
    unsigned ub = __float_as_uint(b);
    ua = (ua + 0x7fffu + ((ua >> 16) & 1u)) >> 16;   // RNE fp32->bf16
    ub = (ub + 0x7fffu + ((ub >> 16) & 1u)) >> 16;
    return ua | (ub << 16);
}

__device__ __forceinline__ unsigned short f32_to_bf16(float a) {
    unsigned ua = __float_as_uint(a);
    ua = (ua + 0x7fffu + ((ua >> 16) & 1u)) >> 16;
    return (unsigned short)ua;
}

// B fragment (32x16 bf16): pre-packed so each lane owns 32 contiguous bytes
__device__ __forceinline__ v16bf b_frag(const unsigned* p) {
    v8u r = *(const v8u*)p;                     // 2x global_load_b128
    return __builtin_bit_cast(v16bf, r);
}

// A fragment (16x32 bf16) from row-major bf16: two contiguous 16B runs per lane
//   lane<16: K in [0,16);  lane>=16: K in [16,32)
__device__ __forceinline__ v16bf a_frag(const unsigned short* ap) {
    uint4 r0 = *(const uint4*)ap;
    uint4 r1 = *(const uint4*)(ap + 8);
    v8u w;
    w[0] = r0.x; w[1] = r0.y; w[2] = r0.z; w[3] = r0.w;
    w[4] = r1.x; w[5] = r1.y; w[6] = r1.z; w[7] = r1.w;
    return __builtin_bit_cast(v16bf, w);
}

__device__ __forceinline__ v8f wmma_bf16(v16bf a, v16bf b, v8f c) {
    return __builtin_amdgcn_wmma_f32_16x16x32_bf16(
        false, a, false, b, (short)0, c, false, false);
}

// sched_group_barrier masks (LLVM AMDGPUIGroupLP)
#define SGB_MFMA      0x0008   // includes WMMA
#define SGB_VMEM_READ 0x0020

// =====================================================================
// Pass 0a: gate_up_w fp32 [E,H,2D] -> bf16 B-fragment-packed W1P
//   frag addr: (((e*KB_A + kb)*NDSUB + dsub)*2 + gu)*256 + lane*8 + v
// =====================================================================
__global__ __launch_bounds__(256)
void pack_gateup_kernel(const float* __restrict__ GUW, unsigned* __restrict__ W1P) {
    unsigned idx = blockIdx.x * 256 + threadIdx.x;   // 0 .. 33,554,431
    int n  = idx & 4095;          // raw column (interleaved gate/up)
    int m  = (idx >> 12) & 15;    // k-pair: k = 2m, 2m+1
    int kb = (idx >> 16) & 63;
    int e  = idx >> 22;

    const float* src = GUW + ((size_t)e * H + kb * 32 + 2 * m) * (2 * D) + n;
    float s0 = __builtin_nontemporal_load(src);             // read-once fp32
    float s1 = __builtin_nontemporal_load(src + 2 * D);

    int dl   = n >> 1;            // d column 0..2047
    int gu   = n & 1;             // 0=gate, 1=up
    int dsub = dl >> 4;
    int nloc = dl & 15;
    int half = (m >= 8) ? 1 : 0;  // B layout: lanes 0-15 K0..15, 16-31 K16..31
    int v    = m - 8 * half;
    int lane = nloc + 16 * half;

    size_t dst = ((((size_t)e * KB_A + kb) * NDSUB + dsub) * 2 + gu) * 256 + lane * 8 + v;
    W1P[dst] = pack_bf16x2(s0, s1);
}

// =====================================================================
// Pass 0b: down_w fp32 [E,D,H] -> bf16 B-fragment-packed W2P
// =====================================================================
__global__ __launch_bounds__(256)
void pack_down_kernel(const float* __restrict__ DW, unsigned* __restrict__ W2P) {
    unsigned idx = blockIdx.x * 256 + threadIdx.x;   // 0 .. 16,777,215
    int n  = idx & 2047;          // h column
    int m  = (idx >> 11) & 15;    // k-pair
    int kb = (idx >> 15) & 63;
    int e  = idx >> 21;

    const float* src = DW + ((size_t)e * D + kb * 32 + 2 * m) * H + n;
    float s0 = __builtin_nontemporal_load(src);
    float s1 = __builtin_nontemporal_load(src + H);

    int hsub = n >> 4;
    int nloc = n & 15;
    int half = (m >= 8) ? 1 : 0;
    int v    = m - 8 * half;
    int lane = nloc + 16 * half;

    size_t dst = (((size_t)e * KB_B + kb) * NHSUB + hsub) * 256 + lane * 8 + v;
    W2P[dst] = pack_bf16x2(s0, s1);
}

// =====================================================================
// Pass 0c: x fp32 [T,H] -> bf16 row-major
// =====================================================================
__global__ __launch_bounds__(256)
void conv_x_kernel(const float* __restrict__ X, unsigned* __restrict__ XB) {
    unsigned idx = blockIdx.x * 256 + threadIdx.x;   // 0 .. T*H/2-1
    const v2f v = __builtin_nontemporal_load((const v2f*)X + idx);
    XB[idx] = pack_bf16x2(v.x, v.y);
}

// =====================================================================
// Kernel A: gate_up GEMM + bias + clamp + GLU  -> fused bf16 [E,T,D]
//   grid (D/256, T/32, E), block 256 = 8 waves
//   wave tile: 16t x 64d (4 d-subtiles, gate+up => 8 accumulators)
//   No LDS, no barriers. sched_group_barrier pins the K-step schedule to
//   [6 loads]([1 wmma][2 loads])x6[1 wmma][1 wmma] so every WMMA overlaps
//   the next fragment's loads instead of s_wait_loadcnt 0 serialization.
// =====================================================================
__global__ __launch_bounds__(256)
void gateup_glu_kernel(const unsigned short* __restrict__ XB,  // [T,H] bf16
                       const unsigned* __restrict__ W1P,       // packed frags
                       const float* __restrict__ GUB,          // [E,2D]
                       unsigned short* __restrict__ FUSED)     // [E,T,D] bf16
{
    const int tid  = threadIdx.x;
    const int lane = tid & 31;
    const int wv   = tid >> 5;
    const int tsub = wv >> 2;                  // 0..1
    const int dq   = wv & 3;                   // 0..3
    const int e    = blockIdx.z;
    const int t0   = blockIdx.y * 32;
    const int dsub0 = blockIdx.x * 16 + dq * 4;   // first of 4 d-subtiles

    const int nloc  = lane & 15;
    const int halfl = lane >> 4;

    v8f acc[8];   // [j*2+0]=gate, [j*2+1]=up for d-subtile j
    #pragma unroll
    for (int j = 0; j < 8; ++j)
        acc[j] = v8f{0.f,0.f,0.f,0.f,0.f,0.f,0.f,0.f};

    const int trow = t0 + tsub * 16 + nloc;
    const unsigned short* arow = XB + (size_t)trow * H + 16 * halfl;

    const size_t kbStride = (size_t)NDSUB * 2 * 256;           // dwords per K-block
    const unsigned* wbase = W1P + (size_t)e * KB_A * kbStride
                                + ((size_t)dsub0 * 2) * 256 + lane * 8;

    for (int kb = 0; kb < KB_A; ++kb) {
        const unsigned* wp = wbase + (size_t)kb * kbStride;
        __builtin_prefetch(wp + kbStride, 0, 0);               // next K-block -> GL2

        v16bf a = a_frag(arow + kb * 32);
        v16bf bfr[8];
        #pragma unroll
        for (int f = 0; f < 8; ++f)
            bfr[f] = b_frag(wp + (size_t)f * 256);

        #pragma unroll
        for (int f = 0; f < 8; ++f)
            acc[f] = wmma_bf16(a, bfr[f], acc[f]);

        // pin schedule: A(2)+b0(2)+b1(2) loads, then (wmma, 2 loads) x6, wmma x2
        __builtin_amdgcn_sched_group_barrier(SGB_VMEM_READ, 6, 0);
        #pragma unroll
        for (int f = 0; f < 6; ++f) {
            __builtin_amdgcn_sched_group_barrier(SGB_MFMA, 1, 0);
            __builtin_amdgcn_sched_group_barrier(SGB_VMEM_READ, 2, 0);
        }
        __builtin_amdgcn_sched_group_barrier(SGB_MFMA, 1, 0);
        __builtin_amdgcn_sched_group_barrier(SGB_MFMA, 1, 0);
    }

    // epilogue: bias + clamp + GLU -> bf16 fused
    const float* B = GUB + (size_t)e * (2 * D);
    #pragma unroll
    for (int j = 0; j < 4; ++j) {
        int d = (dsub0 + j) * 16 + nloc;
        float gb = B[2 * d];
        float ub = B[2 * d + 1];
        #pragma unroll
        for (int r = 0; r < 8; ++r) {
            int t = t0 + tsub * 16 + halfl * 8 + r;
            float g = acc[2 * j][r] + gb;
            g = fminf(g, LIMIT);
            float u = acc[2 * j + 1][r] + ub;
            u = fminf(fmaxf(u, -LIMIT), LIMIT);
            float glu = g / (1.0f + __expf(-ALPHA * g));
            FUSED[((size_t)e * T + t) * D + d] = f32_to_bf16((u + 1.0f) * glu);
        }
    }
}

// =====================================================================
// Kernel B: down GEMM + routing-weighted combine (+down_b) -> fp32 out
//   grid (H/256, T/32), block 256 = 8 waves
//   wave tile: 16t x 64h (4 h-subtiles), experts looped, fp32 accumulate
//   Same sched_group_barrier pipelining: [6]([1][2])x2[1][1].
// =====================================================================
__global__ __launch_bounds__(256)
void down_mix_kernel(const unsigned short* __restrict__ FUSED, // [E,T,D] bf16
                     const unsigned* __restrict__ W2P,         // packed frags
                     const float* __restrict__ DB,             // [E,H]
                     const float* __restrict__ RW,             // [T,E]
                     float* __restrict__ OUT)                  // [T,H]
{
    const int tid  = threadIdx.x;
    const int lane = tid & 31;
    const int wv   = tid >> 5;
    const int tsub = wv >> 2;                   // 0..1
    const int hq   = wv & 3;                    // 0..3
    const int t0   = blockIdx.y * 32;
    const int hsub0 = blockIdx.x * 16 + hq * 4;

    const int nloc  = lane & 15;
    const int halfl = lane >> 4;
    const int trow  = t0 + tsub * 16 + nloc;

    const size_t kbStride = (size_t)NHSUB * 256;   // dwords per K-block

    v8f vout[4];
    #pragma unroll
    for (int j = 0; j < 4; ++j)
        vout[j] = v8f{0.f,0.f,0.f,0.f,0.f,0.f,0.f,0.f};

    for (int e = 0; e < E; ++e) {
        v8f cacc[4];
        #pragma unroll
        for (int j = 0; j < 4; ++j)
            cacc[j] = v8f{0.f,0.f,0.f,0.f,0.f,0.f,0.f,0.f};

        const unsigned short* arow =
            FUSED + ((size_t)e * T + trow) * D + 16 * halfl;
        const unsigned* wbase = W2P + (size_t)e * KB_B * kbStride
                                    + (size_t)hsub0 * 256 + lane * 8;

        for (int kb = 0; kb < KB_B; ++kb) {
            const unsigned* wp = wbase + (size_t)kb * kbStride;
            __builtin_prefetch(wp + kbStride, 0, 0);

            v16bf a = a_frag(arow + kb * 32);
            v16bf bfr[4];
            #pragma unroll
            for (int j = 0; j < 4; ++j)
                bfr[j] = b_frag(wp + (size_t)j * 256);

            #pragma unroll
            for (int j = 0; j < 4; ++j)
                cacc[j] = wmma_bf16(a, bfr[j], cacc[j]);

            __builtin_amdgcn_sched_group_barrier(SGB_VMEM_READ, 6, 0);
            #pragma unroll
            for (int j = 0; j < 2; ++j) {
                __builtin_amdgcn_sched_group_barrier(SGB_MFMA, 1, 0);
                __builtin_amdgcn_sched_group_barrier(SGB_VMEM_READ, 2, 0);
            }
            __builtin_amdgcn_sched_group_barrier(SGB_MFMA, 1, 0);
            __builtin_amdgcn_sched_group_barrier(SGB_MFMA, 1, 0);
        }

        // routed accumulate: out += rw[t,e] * (gemm + down_b[e,h])
        #pragma unroll
        for (int j = 0; j < 4; ++j) {
            int h = (hsub0 + j) * 16 + nloc;
            float dbv = DB[(size_t)e * H + h];
            #pragma unroll
            for (int r = 0; r < 8; ++r) {
                int t = t0 + tsub * 16 + halfl * 8 + r;
                float w = RW[(size_t)t * E + e];
                vout[j][r] += w * (cacc[j][r] + dbv);
            }
        }
    }

    #pragma unroll
    for (int j = 0; j < 4; ++j) {
        int h = (hsub0 + j) * 16 + nloc;
        #pragma unroll
        for (int r = 0; r < 8; ++r) {
            int t = t0 + tsub * 16 + halfl * 8 + r;
            OUT[(size_t)t * H + h] = vout[j][r];
        }
    }
}

// =====================================================================
extern "C" void kernel_launch(void* const* d_in, const int* in_sizes, int n_in,
                              void* d_out, int out_size, void* d_ws, size_t ws_size,
                              hipStream_t stream) {
    const float* x   = (const float*)d_in[0];   // hidden_states [B,S,H]
    const float* rw  = (const float*)d_in[1];   // routing_weights [T,E]
    const float* guw = (const float*)d_in[2];   // gate_up_w [E,H,2D]
    const float* gub = (const float*)d_in[3];   // gate_up_b [E,2D]
    const float* dw  = (const float*)d_in[4];   // down_w [E,D,H]
    const float* db  = (const float*)d_in[5];   // down_b [E,H]

    unsigned* wsd = (unsigned*)d_ws;
    unsigned*       w1p   = wsd + W1P_OFF;
    unsigned*       w2p   = wsd + W2P_OFF;
    unsigned*       xbw   = wsd + XB_OFF;
    unsigned short* xb    = (unsigned short*)xbw;
    unsigned short* fused = (unsigned short*)(wsd + FUSED_OFF);
    float* out = (float*)d_out;

    // Pass 0: fp32 -> bf16 (weights pre-permuted into WMMA fragment order)
    pack_gateup_kernel<<<(unsigned)(W1P_DW / 256), 256, 0, stream>>>(guw, w1p);
    pack_down_kernel  <<<(unsigned)(W2P_DW / 256), 256, 0, stream>>>(dw, w2p);
    conv_x_kernel     <<<(unsigned)(XB_DW  / 256), 256, 0, stream>>>(x, xbw);

    // Pass 1: gate_up + GLU
    dim3 gA(D / 256, T / 32, E);   // (8, 64, 8)
    gateup_glu_kernel<<<gA, 256, 0, stream>>>(xb, w1p, gub, fused);

    // Pass 2: down + routed combine
    dim3 gB(H / 256, T / 32);      // (8, 64)
    down_mix_kernel<<<gB, 256, 0, stream>>>(fused, w2p, db, rw, out);
}
